// Spatial_Attention_layer_57071525429972
// MI455X (gfx1250) — matrix operand order repaired and checked by reading, not compiled, not run on hardware
//
#include <hip/hip_runtime.h>
#include <hip/hip_bf16.h>
#include <math.h>

// ---------------------------------------------------------------------------
// Spatial attention layer, MI455X (gfx1250, wave32, WMMA).
// Math:  M = Y1^T Y3  (Y1 = x@W1 over T, Y3 = x@W3), K = B = 131072
//        out = softmax_axis1( V_s @ sigmoid( W2^T @ M + b_s ) )
// Plan:  kernel 1 (512 WGs): fused Y1/Y3 + bf16 WMMA Gram, split-K partials
//        kernel 2 (32 WGs):  reduce 512 partials -> M (64x64 f32)
//        kernel 3 (1 WG):    60x60 epilogue (2 small matmuls, sigmoid, softmax)
// HBM floor: 157 MB x-read -> ~6.7 us @ 23.3 TB/s; WMMA work ~1 GFLOP (noise).
// ---------------------------------------------------------------------------

#define BATCH   131072
#define TT      5
#define VV      60
#define S_SPLIT 512              // split-K workgroups
#define KC      64               // batches per LDS chunk
#define BPW     (BATCH / S_SPLIT)  // 256 batches per WG -> 4 chunks

typedef __bf16 bf16;
typedef __attribute__((ext_vector_type(16))) __bf16 bf16x16;
typedef __attribute__((ext_vector_type(8)))  __bf16 bf16x8;
typedef __attribute__((ext_vector_type(8)))  float  v8f;

union AB16 { bf16x16 v; bf16x8 h[2]; };

// ---------------------------------------------------------------------------
// Kernel 1: fused Y1/Y3 compute + Gram WMMA over a K-slice of batches.
// 256 threads = 8 waves. Wave w owns tiles (row w>>1, cols 2*(w&1), 2*(w&1)+1)
// of the padded 64x64 output. LDS chunk is stored TRANSPOSED: yT[v][k], rows
// padded to 72 bf16 (36 dwords) -> conflict-free b128 fragment loads.
// ---------------------------------------------------------------------------
__global__ __launch_bounds__(256) void gram_kernel(
    const float* __restrict__ x,
    const float* __restrict__ W1,
    const float* __restrict__ W3,
    float* __restrict__ Mpart)
{
    __shared__ __align__(16) bf16 y1T[64][72];
    __shared__ __align__(16) bf16 y3T[64][72];

    float w1[TT], w3[TT];
#pragma unroll
    for (int t = 0; t < TT; ++t) { w1[t] = W1[t]; w3[t] = W3[t]; }

    const int tid   = threadIdx.x;
    const int wave  = tid >> 5;
    const int lane  = tid & 31;
    const int hlf   = lane >> 4;      // 0: lanes 0-15, 1: lanes 16-31
    const int idx16 = lane & 15;
    const int rt  = wave >> 1;        // tile row 0..3
    const int ct0 = (wave & 1) * 2;   // tile cols ct0, ct0+1

    v8f acc0 = {}; v8f acc1 = {};

    const long long b_base = (long long)blockIdx.x * BPW;

    for (int chunk = 0; chunk < BPW / KC; ++chunk) {
        const long long b0 = b_base + (long long)chunk * KC;

        // Speculative prefetch of next chunk (global_prefetch_b8 path).
        if (chunk + 1 < BPW / KC)
            __builtin_prefetch(x + (b0 + KC) * (TT * VV) + (long long)tid * 75, 0, 1);

        // ---- compute phase: y1/y3 for 64 batches x 60 vertices -------------
        // 3840 (b,v) pairs over 256 threads = 15 each; v varies fastest so the
        // x[] reads are coalesced (v is the contiguous axis of x).
#pragma unroll
        for (int i = 0; i < 15; ++i) {
            const int p  = tid + 256 * i;
            const int bl = p / VV;          // 0..63
            const int v  = p - bl * VV;     // 0..59
            const float* xb = x + (b0 + bl) * (long long)(TT * VV) + v;
            float a1 = 0.f, a3 = 0.f;
#pragma unroll
            for (int t = 0; t < TT; ++t) {
                const float xv = xb[t * VV];
                a1 += xv * w1[t];
                a3 += xv * w3[t];
            }
            y1T[v][bl] = (bf16)a1;   // transposed: row = vertex, col = k
            y3T[v][bl] = (bf16)a3;
        }
        __syncthreads();

        // ---- WMMA phase: 2 k-steps of 32 over this chunk --------------------
#pragma unroll
        for (int ks = 0; ks < 2; ++ks) {
            const int k0 = ks * 32;
            AB16 af, bf0, bf1;
            // A frag (16x32, M x K): element e -> k = k0 + 16*hlf + e
            const bf16* arow = &y1T[16 * rt + idx16][0];
            af.h[0] = *(const bf16x8*)(arow + k0 + 16 * hlf);
            af.h[1] = *(const bf16x8*)(arow + k0 + 16 * hlf + 8);
            // B frag (32x16, K x N): element e -> k = k0 + 16*hlf + e
            const bf16* brow0 = &y3T[16 * ct0 + idx16][0];
            const bf16* brow1 = &y3T[16 * (ct0 + 1) + idx16][0];
            bf0.h[0] = *(const bf16x8*)(brow0 + k0 + 16 * hlf);
            bf0.h[1] = *(const bf16x8*)(brow0 + k0 + 16 * hlf + 8);
            bf1.h[0] = *(const bf16x8*)(brow1 + k0 + 16 * hlf);
            bf1.h[1] = *(const bf16x8*)(brow1 + k0 + 16 * hlf + 8);

            acc0 = __builtin_amdgcn_wmma_f32_16x16x32_bf16(
                false, af.v, false, bf0.v, (short)0, acc0, false, false);
            acc1 = __builtin_amdgcn_wmma_f32_16x16x32_bf16(
                false, af.v, false, bf1.v, (short)0, acc1, false, false);
        }
        __syncthreads();
    }

    // ---- write split-K partial (each WG fully covers the 64x64 tile grid) --
    float* out = Mpart + (size_t)blockIdx.x * 4096;
#pragma unroll
    for (int j = 0; j < 8; ++j) {
        const int row = 16 * rt + j + 8 * hlf;   // C/D layout: VGPR j, half hlf
        out[row * 64 + 16 * ct0 + idx16]       = acc0[j];
        out[row * 64 + 16 * (ct0 + 1) + idx16] = acc1[j];
    }
}

// ---------------------------------------------------------------------------
// Kernel 2: reduce S_SPLIT partials -> M (64x64 f32). 32 blocks x 128 elems.
// ---------------------------------------------------------------------------
__global__ __launch_bounds__(256) void reduce_kernel(
    const float* __restrict__ Mpart, float* __restrict__ M)
{
    __shared__ float sdata[256];
    const int t = threadIdx.x;
    const int e = blockIdx.x * 128 + (t & 127);
    const int s0 = (t >> 7) * (S_SPLIT / 2);
    float acc = 0.f;
    for (int s = 0; s < S_SPLIT / 2; ++s)
        acc += Mpart[(size_t)(s0 + s) * 4096 + e];
    sdata[t] = acc;
    __syncthreads();
    if (t < 128)
        M[blockIdx.x * 128 + t] = sdata[t] + sdata[t + 128];
}

// ---------------------------------------------------------------------------
// Kernel 3: epilogue.  P = sigmoid(W2^T M + b_s); S = V_s P; softmax over axis 1.
// ---------------------------------------------------------------------------
__global__ __launch_bounds__(256) void finish_kernel(
    const float* __restrict__ M,
    const float* __restrict__ W2,
    const float* __restrict__ b_s,
    const float* __restrict__ V_s,
    float* __restrict__ out)
{
    __shared__ float Ms[VV][VV + 1];
    __shared__ float Ps[VV][VV + 1];
    __shared__ float Ss[VV][VV + 1];
    const int t = threadIdx.x;

    for (int idx = t; idx < VV * VV; idx += 256) {
        const int i = idx / VV, j = idx - i * VV;
        Ms[i][j] = M[i * 64 + j];                    // M stored with stride 64
    }
    __syncthreads();

    for (int idx = t; idx < VV * VV; idx += 256) {   // product = W2^T @ M, +b, sigmoid
        const int i = idx / VV, j = idx - i * VV;
        float acc = 0.f;
        for (int v = 0; v < VV; ++v) acc += W2[v * VV + i] * Ms[v][j];
        const float z = acc + b_s[idx];
        Ps[i][j] = 1.0f / (1.0f + __expf(-z));
    }
    __syncthreads();

    for (int idx = t; idx < VV * VV; idx += 256) {   // S = V_s @ sigmoid(...)
        const int i = idx / VV, j = idx - i * VV;
        float acc = 0.f;
        for (int v = 0; v < VV; ++v) acc += V_s[i * VV + v] * Ps[v][j];
        Ss[i][j] = acc;
    }
    __syncthreads();

    if (t < VV) {                                    // softmax over axis 1 (rows i), per column j
        const int j = t;
        float mx = Ss[0][j];
        for (int i = 1; i < VV; ++i) mx = fmaxf(mx, Ss[i][j]);
        float sum = 0.f;
        for (int i = 0; i < VV; ++i) { const float e = __expf(Ss[i][j] - mx); Ss[i][j] = e; sum += e; }
        const float inv = 1.0f / sum;
        for (int i = 0; i < VV; ++i) out[i * VV + j] = Ss[i][j] * inv;
    }
}

// ---------------------------------------------------------------------------
extern "C" void kernel_launch(void* const* d_in, const int* in_sizes, int n_in,
                              void* d_out, int out_size, void* d_ws, size_t ws_size,
                              hipStream_t stream)
{
    const float* x   = (const float*)d_in[0];
    const float* W1  = (const float*)d_in[1];
    const float* W2  = (const float*)d_in[2];
    const float* W3  = (const float*)d_in[3];
    const float* b_s = (const float*)d_in[4];
    const float* V_s = (const float*)d_in[5];

    float* Mpart = (float*)d_ws;                         // S_SPLIT * 4096 f32 (8 MB)
    float* M     = Mpart + (size_t)S_SPLIT * 4096;       // 4096 f32

    gram_kernel  <<<S_SPLIT, 256, 0, stream>>>(x, W1, W3, Mpart);
    reduce_kernel<<<32,      256, 0, stream>>>(Mpart, M);
    finish_kernel<<<1,       256, 0, stream>>>(M, W2, b_s, V_s, (float*)d_out);
}
